// MultiHeadAttention_30863634989327
// MI455X (gfx1250) — matrix operand order repaired and checked by reading, not compiled, not run on hardware
//
#include <hip/hip_runtime.h>
#include <hip/hip_bf16.h>

// ---------------------------------------------------------------------------
// MHA forward for MI455X (gfx1250, wave32, WMMA).
// D_MODEL=1024, H=16, Dk=64, B=2, T=2048  ->  M = B*T = 4096 flattened rows.
// All matmuls run through v_wmma_f32_16x16x32_f16 (f32 accumulate).
// Double-buffered LDS staging; async global->LDS copies where layout allows.
// ---------------------------------------------------------------------------

typedef __attribute__((ext_vector_type(16))) _Float16 v16h;
typedef __attribute__((ext_vector_type(8)))  _Float16 v8h;
typedef __attribute__((ext_vector_type(8)))  float    v8f;
typedef int v4i_t __attribute__((vector_size(16)));  // 128-bit payload type

#define WMMA16(a, b, c) \
  __builtin_amdgcn_wmma_f32_16x16x32_f16(false, (a), false, (b), (short)0, (c), false, false)

#if __has_builtin(__builtin_amdgcn_global_load_async_to_lds_b128) && \
    __has_builtin(__builtin_amdgcn_s_wait_asynccnt)
#define HAVE_ASYNC_LDS 1
#else
#define HAVE_ASYNC_LDS 0
#endif

static __device__ __forceinline__ v16h mk16(v8h lo, v8h hi) {
  v16h r;
#pragma unroll
  for (int i = 0; i < 8; ++i) { r[i] = lo[i]; r[i + 8] = hi[i]; }
  return r;
}

// 16-lane butterfly reductions (masks 8..1 never cross a 16-lane half, which
// matches the WMMA C-layout row mapping M = j + 8*(lane/16)).
static __device__ __forceinline__ float redmax16(float v) {
#pragma unroll
  for (int m = 8; m >= 1; m >>= 1) v = fmaxf(v, __shfl_xor(v, m, 32));
  return v;
}
static __device__ __forceinline__ float redsum16(float v) {
#pragma unroll
  for (int m = 8; m >= 1; m >>= 1) v += __shfl_xor(v, m, 32);
  return v;
}

// ---------------------------------------------------------------------------
// C[m][n] = sum_k A[m][k] * W[n][k] + bias[n]
//   A: [4096 x 1024] fp32 (A_HALF=0) or f16 (A_HALF=1)
//   W: [1024 x 1024] fp32 row-major (torch Linear weight: out x in)
// Block = 256 threads = 8 waves; tile = 128(M) x 64(N); k-step = 32.
// Double-buffered LDS; loop order = issue(next) -> WMMA(cur) -> commit(next).
// ---------------------------------------------------------------------------
template <int A_HALF, int OUT_FLOAT>
__global__ void __launch_bounds__(256)
gemm_bias_f16wmma(const void* __restrict__ A_, const float* __restrict__ W,
                  const float* __restrict__ bias, void* __restrict__ outp) {
  __shared__ _Float16 a_s[2][128][32];  // 2 x 8 KB
  __shared__ _Float16 b_s[2][64][32];   // 2 x 4 KB

  const int tid  = threadIdx.x;
  const int lane = tid & 31, w = tid >> 5;
  const int l16  = lane & 15, half = lane >> 4;
  const int m0   = blockIdx.x * 128;
  const int n0   = blockIdx.y * 64;

  const float*    Af = (const float*)A_;
  const _Float16* Ah = (const _Float16*)A_;

  // staging assignment: A -> (row, 16-col half), B -> (row, 8-col quarter)
  const int arow = tid >> 1, acs = (tid & 1) * 16;
  const int brow = tid >> 2, bcs = (tid & 3) * 8;

  float4 ra[4];   // A tile regs (fp32 path)
  v8h    rah[2];  // A tile regs (f16 path)
  float4 rb[2];   // B tile regs

#define GEMM_ISSUE(K0)                                                        \
  do {                                                                        \
    if (A_HALF) {                                                             \
      const _Float16* s = Ah + (size_t)(m0 + arow) * 1024 + (K0) + acs;       \
      rah[0] = *(const v8h*)s; rah[1] = *(const v8h*)(s + 8);                 \
    } else {                                                                  \
      const float* s = Af + (size_t)(m0 + arow) * 1024 + (K0) + acs;          \
      ra[0] = *(const float4*)(s + 0);  ra[1] = *(const float4*)(s + 4);      \
      ra[2] = *(const float4*)(s + 8);  ra[3] = *(const float4*)(s + 12);     \
    }                                                                         \
    const float* sb = W + (size_t)(n0 + brow) * 1024 + (K0) + bcs;            \
    rb[0] = *(const float4*)sb; rb[1] = *(const float4*)(sb + 4);             \
  } while (0)

#define GEMM_COMMIT(BUF)                                                      \
  do {                                                                        \
    _Float16* d = &a_s[(BUF)][arow][acs];                                     \
    if (A_HALF) {                                                             \
      *(v8h*)d = rah[0]; *(v8h*)(d + 8) = rah[1];                             \
    } else {                                                                  \
      d[0]=(_Float16)ra[0].x;  d[1]=(_Float16)ra[0].y;                        \
      d[2]=(_Float16)ra[0].z;  d[3]=(_Float16)ra[0].w;                        \
      d[4]=(_Float16)ra[1].x;  d[5]=(_Float16)ra[1].y;                        \
      d[6]=(_Float16)ra[1].z;  d[7]=(_Float16)ra[1].w;                        \
      d[8]=(_Float16)ra[2].x;  d[9]=(_Float16)ra[2].y;                        \
      d[10]=(_Float16)ra[2].z; d[11]=(_Float16)ra[2].w;                       \
      d[12]=(_Float16)ra[3].x; d[13]=(_Float16)ra[3].y;                       \
      d[14]=(_Float16)ra[3].z; d[15]=(_Float16)ra[3].w;                       \
    }                                                                         \
    _Float16* db = &b_s[(BUF)][brow][bcs];                                    \
    db[0]=(_Float16)rb[0].x; db[1]=(_Float16)rb[0].y;                         \
    db[2]=(_Float16)rb[0].z; db[3]=(_Float16)rb[0].w;                         \
    db[4]=(_Float16)rb[1].x; db[5]=(_Float16)rb[1].y;                         \
    db[6]=(_Float16)rb[1].z; db[7]=(_Float16)rb[1].w;                         \
  } while (0)

  v8f acc[4] = {};

  GEMM_ISSUE(0);
  GEMM_COMMIT(0);
  __syncthreads();

  int p = 0;
  for (int k0 = 0; k0 < 1024; k0 += 32) {
    const bool more = (k0 + 32) < 1024;
    if (more) GEMM_ISSUE(k0 + 32);  // global loads overlap WMMAs below

    const int ar  = w * 16 + l16;
    const int kbo = half * 16;
    v16h a = mk16(*(const v8h*)&a_s[p][ar][kbo],
                  *(const v8h*)&a_s[p][ar][kbo + 8]);
    v16h bfr[4];
#pragma unroll
    for (int nt = 0; nt < 4; ++nt)
      bfr[nt] = mk16(*(const v8h*)&b_s[p][nt * 16 + l16][half * 16],
                     *(const v8h*)&b_s[p][nt * 16 + l16][half * 16 + 8]);
#pragma unroll
    for (int nt = 0; nt < 4; ++nt) acc[nt] = WMMA16(a, bfr[nt], acc[nt]);

    if (more) GEMM_COMMIT(p ^ 1);   // s_wait_loadcnt lands here, after WMMAs
    __syncthreads();
    p ^= 1;
  }
#undef GEMM_ISSUE
#undef GEMM_COMMIT

  // ---- epilogue: bias + store (C layout: lane col = l16, rows j+8*half) ---
#pragma unroll
  for (int nt = 0; nt < 4; ++nt) {
    const int n = n0 + nt * 16 + l16;
    const float bv = bias[n];
#pragma unroll
    for (int j = 0; j < 8; ++j) {
      const int m = m0 + w * 16 + j + half * 8;
      const float val = acc[nt][j] + bv;
      if (OUT_FLOAT) ((float*)outp)[(size_t)m * 1024 + n] = val;
      else           ((_Float16*)outp)[(size_t)m * 1024 + n] = (_Float16)val;
    }
  }
}

// ---------------------------------------------------------------------------
// Flash attention: one block per (b, h, 128-query slab). 8 waves, each wave
// owns 16 query rows (q fragments + 16x64 f32 accumulator in registers).
// 32-key K/V tiles stream through double-buffered LDS; K tiles use async
// global->LDS copies (ASYNCcnt) when available; online softmax keeps the
// 2048x2048 score matrix entirely on chip.
// ---------------------------------------------------------------------------
__global__ void __launch_bounds__(256)
mha_flash_f16wmma(const _Float16* __restrict__ qb, const _Float16* __restrict__ kb,
                  const _Float16* __restrict__ vb, _Float16* __restrict__ ctx) {
  __shared__ _Float16 k_s[2][32][64];    // 2 x 4 KB : K tile, row-major [key][d]
  __shared__ _Float16 vt_s[2][64][32];   // 2 x 4 KB : V tile, transposed [d][key]
  __shared__ _Float16 p_s[8][16][32];    // 8 KB : per-wave P staging (C->A relayout)

  const int tid  = threadIdx.x;
  const int lane = tid & 31, w = tid >> 5;
  const int l16  = lane & 15, half = lane >> 4;
  const int qblk = blockIdx.x;  // 0..15  (128 queries each)
  const int h    = blockIdx.y;  // 0..15
  const int b    = blockIdx.z;  // 0..1
  const size_t base = (size_t)b * 2048 * 1024 + (size_t)h * 64;

  // q A-fragments for this wave's 16 rows (row = l16), d = 0..31 and 32..63
  const _Float16* qrow = qb + base + (size_t)(qblk * 128 + w * 16 + l16) * 1024;
  const int kb8 = half * 16;
  const v16h aq0 = mk16(*(const v8h*)(qrow + kb8),      *(const v8h*)(qrow + kb8 + 8));
  const v16h aq1 = mk16(*(const v8h*)(qrow + 32 + kb8), *(const v8h*)(qrow + 32 + kb8 + 8));

  v8f o[4] = {};
  float rmax[8], rsum[8];
#pragma unroll
  for (int j = 0; j < 8; ++j) { rmax[j] = -1e30f; rsum[j] = 0.0f; }

  const int key  = tid >> 3;         // 0..31  (cooperative tile loads)
  const int dseg = (tid & 7) * 8;    // 0..56

#if HAVE_ASYNC_LDS
#define STAGE_K(KT, BUF)                                                      \
  __builtin_amdgcn_global_load_async_to_lds_b128(                             \
      (__attribute__((address_space(1))) v4i_t*)(void*)                       \
          (kb + base + (size_t)((KT) * 32 + key) * 1024 + dseg),              \
      (__attribute__((address_space(3))) v4i_t*)(void*)                       \
          (&k_s[(BUF)][key][dseg]), 0, 0)
#define WAIT_ASYNC() __builtin_amdgcn_s_wait_asynccnt(0)
#else
#define STAGE_K(KT, BUF)                                                      \
  (*(v8h*)&k_s[(BUF)][key][dseg] =                                            \
       *(const v8h*)(kb + base + (size_t)((KT) * 32 + key) * 1024 + dseg))
#define WAIT_ASYNC() ((void)0)
#endif

#define ISSUE_V(KT) (vv = *(const v8h*)(vb + base + (size_t)((KT) * 32 + key) * 1024 + dseg))
#define COMMIT_V(BUF)                                                         \
  do {                                                                        \
    _Float16* vd = &vt_s[(BUF)][dseg][key];                                   \
    _Pragma("unroll")                                                         \
    for (int i = 0; i < 8; ++i) vd[(size_t)i * 32] = vv[i];                   \
  } while (0)

  v8h vv;

  // prologue: stage tile 0 into buffer 0
  STAGE_K(0, 0);
  ISSUE_V(0);
  COMMIT_V(0);
  WAIT_ASYNC();
  __syncthreads();

  int p = 0;
  for (int kt = 0; kt < 64; ++kt) {
    const bool more = (kt + 1) < 64;
    if (more) {
      STAGE_K(kt + 1, p ^ 1);     // async copy runs under the WMMAs below
      ISSUE_V(kt + 1);            // global load; wait lands after compute
      if (kt + 2 < 64) {          // prefetch tile after next
        __builtin_prefetch(kb + base + (size_t)((kt + 2) * 32 + key) * 1024 + dseg, 0, 3);
        __builtin_prefetch(vb + base + (size_t)((kt + 2) * 32 + key) * 1024 + dseg, 0, 3);
      }
    }

    // ---- S = q . k^T  (two 16x16 key tiles, K-dim = 64 in two WMMA steps) -
    v8f S0 = {}, S1 = {};
    {
      v16h b00 = mk16(*(const v8h*)&k_s[p][l16][half * 16],
                      *(const v8h*)&k_s[p][l16][half * 16 + 8]);
      v16h b01 = mk16(*(const v8h*)&k_s[p][l16][32 + half * 16],
                      *(const v8h*)&k_s[p][l16][32 + half * 16 + 8]);
      v16h b10 = mk16(*(const v8h*)&k_s[p][16 + l16][half * 16],
                      *(const v8h*)&k_s[p][16 + l16][half * 16 + 8]);
      v16h b11 = mk16(*(const v8h*)&k_s[p][16 + l16][32 + half * 16],
                      *(const v8h*)&k_s[p][16 + l16][32 + half * 16 + 8]);
      S0 = WMMA16(aq0, b00, S0);
      S0 = WMMA16(aq1, b01, S0);
      S1 = WMMA16(aq0, b10, S1);
      S1 = WMMA16(aq1, b11, S1);
    }

    // ---- online softmax (per row j of this wave's 16 rows) ----------------
    const float sc = 0.125f;  // 1/sqrt(Dk)
#pragma unroll
    for (int j = 0; j < 8; ++j) {
      const float x0 = S0[j] * sc, x1 = S1[j] * sc;
      float mx = redmax16(fmaxf(x0, x1));
      mx = fmaxf(mx, rmax[j]);
      const float corr = __expf(rmax[j] - mx);
      rmax[j] = mx;
      const float p0 = __expf(x0 - mx), p1 = __expf(x1 - mx);
      rsum[j] = rsum[j] * corr + redsum16(p0 + p1);
      o[0][j] *= corr; o[1][j] *= corr; o[2][j] *= corr; o[3][j] *= corr;
      p_s[w][j + half * 8][l16]      = (_Float16)p0;   // relayout C -> A
      p_s[w][j + half * 8][16 + l16] = (_Float16)p1;
    }
    asm volatile("s_wait_dscnt 0" ::: "memory");  // wave-local P RAW through LDS

    // ---- O += P . V  (P is 16x32 A-fragment, V gives 4 N-tiles of 16) -----
    v16h ap = mk16(*(const v8h*)&p_s[w][l16][kb8],
                   *(const v8h*)&p_s[w][l16][kb8 + 8]);
#pragma unroll
    for (int nt = 0; nt < 4; ++nt) {
      v16h bvf = mk16(*(const v8h*)&vt_s[p][nt * 16 + l16][half * 16],
                      *(const v8h*)&vt_s[p][nt * 16 + l16][half * 16 + 8]);
      o[nt] = WMMA16(ap, bvf, o[nt]);
    }

    if (more) COMMIT_V(p ^ 1);  // s_wait_loadcnt for V lands here, post-WMMA
    WAIT_ASYNC();               // next K tile resident before the barrier
    __syncthreads();
    p ^= 1;
  }
#undef STAGE_K
#undef WAIT_ASYNC
#undef ISSUE_V
#undef COMMIT_V

  // ---- normalize and write context (f16, [B,T,1024] with head slice) ------
#pragma unroll
  for (int nt = 0; nt < 4; ++nt) {
#pragma unroll
    for (int j = 0; j < 8; ++j) {
      const int t = qblk * 128 + w * 16 + j + half * 8;
      ctx[base + (size_t)t * 1024 + nt * 16 + l16] = (_Float16)(o[nt][j] / rsum[j]);
    }
  }
}

// ---------------------------------------------------------------------------
extern "C" void kernel_launch(void* const* d_in, const int* in_sizes, int n_in,
                              void* d_out, int out_size, void* d_ws, size_t ws_size,
                              hipStream_t stream) {
  const float* Q  = (const float*)d_in[0];
  const float* K  = (const float*)d_in[1];
  const float* V  = (const float*)d_in[2];
  const float* Wq = (const float*)d_in[3];
  const float* bq = (const float*)d_in[4];
  const float* Wk = (const float*)d_in[5];
  const float* bk = (const float*)d_in[6];
  const float* Wv = (const float*)d_in[7];
  const float* bv = (const float*)d_in[8];
  const float* Wo = (const float*)d_in[9];
  const float* bo = (const float*)d_in[10];

  const size_t MN = (size_t)4096 * 1024;  // B*T x D_MODEL
  _Float16* qb   = (_Float16*)d_ws;       // 8 MB each, 32 MB total
  _Float16* kbuf = qb + MN;
  _Float16* vbuf = kbuf + MN;
  _Float16* ctx  = vbuf + MN;

  dim3 gg(32, 16);  // 4096/128 x 1024/64
  gemm_bias_f16wmma<0, 0><<<gg, 256, 0, stream>>>(Q, Wq, bq, qb);
  gemm_bias_f16wmma<0, 0><<<gg, 256, 0, stream>>>(K, Wk, bk, kbuf);
  gemm_bias_f16wmma<0, 0><<<gg, 256, 0, stream>>>(V, Wv, bv, vbuf);

  mha_flash_f16wmma<<<dim3(16, 16, 2), 256, 0, stream>>>(qb, kbuf, vbuf, ctx);

  gemm_bias_f16wmma<1, 1><<<gg, 256, 0, stream>>>(ctx, Wo, bo, d_out);
}
